// CCEdgeGuide_61220463837597
// MI455X (gfx1250) — matrix operand order repaired and checked by reading, not compiled, not run on hardware
//
#include <hip/hip_runtime.h>

// CCEdgeGuide on MI455X (gfx1250, wave32).
// Attention weights computed once (max_edge cancels in softmax), stored f16
// (113MB total -> L2 resident), aggregation iterations as WMMA f16 GEMMs
// (M=K=192, N=32 padded) with f32 accumulation.

static constexpr int Bn = 4, Cn = 19, Hn = 192, Wn = 192, CP = 32;
static constexpr int LST = 200;               // LDS f16 stride per channel row (16B-aligned rows)
static constexpr float THETA_ = 40.0f;

typedef __attribute__((ext_vector_type(16))) _Float16 v16h;
typedef __attribute__((ext_vector_type(8)))  float    v8f;
typedef __attribute__((ext_vector_type(2)))  _Float16 h2;

union Frag  { v16h v; uint4 q[2]; };
union U32H2 { unsigned u; h2 h; };

// ---------------- cumsums of relu(edge) ----------------
__global__ void k_cumsum(const float* __restrict__ edge,
                         float* __restrict__ hc, float* __restrict__ wc) {
  int b = blockIdx.x % Bn;
  int mode = blockIdx.x / Bn;
  const float* e = edge + (size_t)b * Hn * Wn;
  if (mode == 0) {                       // hc: cumsum over i, thread = j (coalesced)
    int j = threadIdx.x;
    float r = 0.f;
    for (int i = 0; i < Hn; ++i) {
      float v = e[i * Wn + j];
      r += v > 0.f ? v : 0.f;
      hc[(size_t)b * Hn * Wn + i * Wn + j] = r;
    }
  } else {                               // wc: cumsum over j, thread = i
    int i = threadIdx.x;
    float r = 0.f;
    for (int j = 0; j < Wn; ++j) {
      float v = e[i * Wn + j];
      r += v > 0.f ? v : 0.f;
      wc[(size_t)b * Hn * Wn + i * Wn + j] = r;
    }
  }
}

// ---------------- column attention + 1/Z ----------------
// block = (b, j), thread = i.  Acol[b][j][i][t] = softmax weight of (t,j) for query (i,j)
__global__ void k_attn_col(const float* __restrict__ hc, const float* __restrict__ wc,
                           float* __restrict__ zinv, _Float16* __restrict__ Acol) {
  int b = blockIdx.x / Wn, j = blockIdx.x % Wn, i = threadIdx.x;
  __shared__ float hcol[Hn];
  hcol[i] = hc[(size_t)b * Hn * Wn + i * Wn + j];
  __syncthreads();
  float ci = hcol[i];
  const float* wr = wc + (size_t)b * Hn * Wn + (size_t)i * Wn;
  float cj = wr[j];
  float s_row = 0.f, s_col = 0.f;
  for (int s = 0; s < Wn; ++s) s_row += __expf(-THETA_ * fabsf(wr[s] - cj));
  for (int t = 0; t < Hn; ++t) s_col += __expf(-THETA_ * fabsf(hcol[t] - ci));
  float inv = 1.0f / (s_col + s_row - 1.0f);     // exclude s==j (exp(0)=1)
  zinv[(size_t)(b * Hn + i) * Wn + j] = inv;
  _Float16* out = Acol + ((size_t)(b * Wn + j) * Hn + i) * Hn;
  for (int t = 0; t < Hn; t += 2) {
    h2 p;
    p[0] = (_Float16)(__expf(-THETA_ * fabsf(hcol[t]     - ci)) * inv);
    p[1] = (_Float16)(__expf(-THETA_ * fabsf(hcol[t + 1] - ci)) * inv);
    *(h2*)(out + t) = p;
  }
}

// ---------------- row attention (diag zero) ----------------
// block = (b, i), thread = j.  Arow[b][i][j][s]
__global__ void k_attn_row(const float* __restrict__ wc, const float* __restrict__ zinv,
                           _Float16* __restrict__ Arow) {
  int b = blockIdx.x / Hn, i = blockIdx.x % Hn, j = threadIdx.x;
  __shared__ float wrow[Wn];
  wrow[j] = wc[(size_t)b * Hn * Wn + i * Wn + j];
  __syncthreads();
  float cj = wrow[j];
  float inv = zinv[(size_t)(b * Hn + i) * Wn + j];
  _Float16* out = Arow + ((size_t)(b * Hn + i) * Wn + j) * Wn;
  for (int s = 0; s < Wn; s += 2) {
    float a0 = (s     == j) ? 0.f : __expf(-THETA_ * fabsf(wrow[s]     - cj)) * inv;
    float a1 = (s + 1 == j) ? 0.f : __expf(-THETA_ * fabsf(wrow[s + 1] - cj)) * inv;
    h2 p; p[0] = (_Float16)a0; p[1] = (_Float16)a1;
    *(h2*)(out + s) = p;
  }
}

// ---------------- pack mask -> f16 transposed/padded [b][i][j][32] ----------------
__global__ void k_pack(const float* __restrict__ mask, _Float16* __restrict__ featp) {
  int tid = blockIdx.x * blockDim.x + threadIdx.x;
  if (tid >= Bn * Hn * Wn) return;
  int b = tid / (Hn * Wn);
  int rem = tid - b * Hn * Wn;
  const float* mp = mask + (size_t)b * Cn * Hn * Wn + rem;
  _Float16* op = featp + (size_t)tid * CP;
  for (int c = 0; c < CP; c += 2) {
    h2 p;
    p[0] = (c     < Cn) ? (_Float16)mp[(size_t)c       * Hn * Wn] : (_Float16)0.f;
    p[1] = (c + 1 < Cn) ? (_Float16)mp[(size_t)(c + 1) * Hn * Wn] : (_Float16)0.f;
    *(h2*)(op + c) = p;
  }
}

// ---------------- GEMM (column pass): T1[b][i][j][c] = sum_t Acol(i,t)*feat[b][t][j][c] ----------------
// block = (b, j), 12 waves, wave w owns M strip [16w,16w+16)
__global__ void __launch_bounds__(384) k_gemm_col(const _Float16* __restrict__ Acol,
                                                  const _Float16* __restrict__ feat,
                                                  float* __restrict__ T1) {
  int b = blockIdx.x / Wn, j = blockIdx.x % Wn;
  int tid = threadIdx.x;
  __shared__ _Float16 ldsT[CP * LST];           // [c][t], transposed so B-frags are contiguous
  {
    int c2 = tid & 15, tr = tid >> 4;           // 24 t-rows per pass
    for (int t = tr; t < Hn; t += 24) {
      const unsigned* src = (const unsigned*)(feat + ((size_t)(b * Hn + t) * Wn + j) * CP);
      U32H2 u; u.u = src[c2];
      ldsT[(2 * c2)     * LST + t] = u.h[0];
      ldsT[(2 * c2 + 1) * LST + t] = u.h[1];
    }
  }
  __syncthreads();
  int lane = tid & 31, wv = tid >> 5;
  int hf = lane >> 4, n15 = lane & 15;
  const _Float16* arow = Acol + ((size_t)(b * Wn + j) * Hn + (wv * 16 + n15)) * Hn;
  v8f acc0 = {}; v8f acc1 = {};
#pragma unroll
  for (int kt = 0; kt < 6; ++kt) {
    Frag a, b0, b1;
    // A 16x32 f16 layout: lane row = lane&15; e<8 -> K = 32kt+8*hf+e ; e>=8 -> K = 32kt+16+8*hf+(e-8)
    a.q[0] = *(const uint4*)(arow + kt * 32 + hf * 8);
    a.q[1] = *(const uint4*)(arow + kt * 32 + 16 + hf * 8);
    int t0 = kt * 32 + hf * 8;                  // B: lane col = lane&15, same K order as A
    b0.q[0] = *(const uint4*)(&ldsT[n15        * LST + t0]);
    b0.q[1] = *(const uint4*)(&ldsT[n15        * LST + t0 + 16]);
    b1.q[0] = *(const uint4*)(&ldsT[(16 + n15) * LST + t0]);
    b1.q[1] = *(const uint4*)(&ldsT[(16 + n15) * LST + t0 + 16]);
    acc0 = __builtin_amdgcn_wmma_f32_16x16x32_f16(false, a.v, false, b0.v, (short)0, acc0, false, false);
    acc1 = __builtin_amdgcn_wmma_f32_16x16x32_f16(false, a.v, false, b1.v, (short)0, acc1, false, false);
  }
#pragma unroll
  for (int v = 0; v < 8; ++v) {                 // C/D: vgpr v -> M = v + 8*hf, N = lane&15
    int i = wv * 16 + v + 8 * hf;
    size_t o = ((size_t)(b * Hn + i) * Wn + j) * CP;
    T1[o + n15]      = acc0[v];
    T1[o + 16 + n15] = acc1[v];
  }
}

// ---------------- GEMM (row pass) + fuse add: next = T1 + sum_s Arow(j,s)*feat[b][i][s][c] ----------------
// block = (b, i)
__global__ void __launch_bounds__(384) k_gemm_row(const _Float16* __restrict__ Arow,
                                                  const _Float16* __restrict__ feat,
                                                  const float* __restrict__ T1,
                                                  _Float16* __restrict__ featNext,
                                                  float* __restrict__ T2) {
  int b = blockIdx.x / Hn, i = blockIdx.x % Hn;
  int tid = threadIdx.x;
  __shared__ _Float16 ldsT[CP * LST];
  {
    int c2 = tid & 15, sr = tid >> 4;
    const unsigned* base = (const unsigned*)(feat + ((size_t)(b * Hn + i) * Wn) * CP);
    for (int s = sr; s < Wn; s += 24) {
      U32H2 u; u.u = base[s * (CP / 2) + c2];
      ldsT[(2 * c2)     * LST + s] = u.h[0];
      ldsT[(2 * c2 + 1) * LST + s] = u.h[1];
    }
  }
  __syncthreads();
  int lane = tid & 31, wv = tid >> 5;
  int hf = lane >> 4, n15 = lane & 15;
  const _Float16* arow = Arow + ((size_t)(b * Hn + i) * Wn + (wv * 16 + n15)) * Wn;
  v8f acc0 = {}; v8f acc1 = {};
#pragma unroll
  for (int kt = 0; kt < 6; ++kt) {
    Frag a, b0, b1;
    a.q[0] = *(const uint4*)(arow + kt * 32 + hf * 8);
    a.q[1] = *(const uint4*)(arow + kt * 32 + 16 + hf * 8);
    int t0 = kt * 32 + hf * 8;
    b0.q[0] = *(const uint4*)(&ldsT[n15        * LST + t0]);
    b0.q[1] = *(const uint4*)(&ldsT[n15        * LST + t0 + 16]);
    b1.q[0] = *(const uint4*)(&ldsT[(16 + n15) * LST + t0]);
    b1.q[1] = *(const uint4*)(&ldsT[(16 + n15) * LST + t0 + 16]);
    acc0 = __builtin_amdgcn_wmma_f32_16x16x32_f16(false, a.v, false, b0.v, (short)0, acc0, false, false);
    acc1 = __builtin_amdgcn_wmma_f32_16x16x32_f16(false, a.v, false, b1.v, (short)0, acc1, false, false);
  }
#pragma unroll
  for (int v = 0; v < 8; ++v) {
    int j = wv * 16 + v + 8 * hf;
    size_t o = ((size_t)(b * Hn + i) * Wn + j) * CP;
    float r0 = acc0[v] + T1[o + n15];
    float r1 = acc1[v] + T1[o + 16 + n15];
    featNext[o + n15]      = (_Float16)r0;
    featNext[o + 16 + n15] = (_Float16)r1;
    T2[o + n15]      = r0;
    T2[o + 16 + n15] = r1;
  }
}

// ---------------- emit f32 output [b][c][i][j] ----------------
__global__ void k_emit(const float* __restrict__ T2, float* __restrict__ out) {
  int tid = blockIdx.x * blockDim.x + threadIdx.x;
  if (tid >= Bn * Cn * Hn * Wn) return;
  int j = tid % Wn;
  int t = tid / Wn;
  int i = t % Hn; t /= Hn;
  int c = t % Cn; int b = t / Cn;
  out[tid] = T2[((size_t)(b * Hn + i) * Wn + j) * CP + c];
}

extern "C" void kernel_launch(void* const* d_in, const int* in_sizes, int n_in,
                              void* d_out, int out_size, void* d_ws, size_t ws_size,
                              hipStream_t stream) {
  (void)in_sizes; (void)n_in; (void)out_size; (void)ws_size;
  const float* mask = (const float*)d_in[0];
  const float* edge = (const float*)d_in[1];
  // d_in[2] is `iter` on device; reference fixes iter=3 and we cannot sync-read
  // a device scalar during graph capture -> hardcoded 3 iterations.

  char* ws = (char*)d_ws;
  auto carve = [&](size_t bytes) -> char* {
    char* p = ws; ws += (bytes + 255) & ~(size_t)255; return p;
  };
  const size_t HW   = (size_t)Hn * Wn;
  float*     hc   = (float*)    carve(Bn * HW * sizeof(float));
  float*     wc   = (float*)    carve(Bn * HW * sizeof(float));
  float*     zinv = (float*)    carve(Bn * HW * sizeof(float));
  _Float16*  Acol = (_Float16*) carve((size_t)Bn * Wn * Hn * Hn * sizeof(_Float16));
  _Float16*  Arw  = (_Float16*) carve((size_t)Bn * Hn * Wn * Wn * sizeof(_Float16));
  _Float16*  fA   = (_Float16*) carve(Bn * HW * CP * sizeof(_Float16));
  _Float16*  fB   = (_Float16*) carve(Bn * HW * CP * sizeof(_Float16));
  float*     T1   = (float*)    carve(Bn * HW * CP * sizeof(float));
  float*     T2   = (float*)    carve(Bn * HW * CP * sizeof(float));

  k_cumsum  <<<dim3(Bn * 2),  dim3(Wn),  0, stream>>>(edge, hc, wc);
  k_attn_col<<<dim3(Bn * Wn), dim3(Hn),  0, stream>>>(hc, wc, zinv, Acol);
  k_attn_row<<<dim3(Bn * Hn), dim3(Wn),  0, stream>>>(wc, zinv, Arw);
  k_pack    <<<dim3((Bn * Hn * Wn + 255) / 256), dim3(256), 0, stream>>>(mask, fA);

  _Float16* cur = fA; _Float16* nxt = fB;
  for (int it = 0; it < 3; ++it) {
    k_gemm_col<<<dim3(Bn * Wn), dim3(384), 0, stream>>>(Acol, cur, T1);
    k_gemm_row<<<dim3(Bn * Hn), dim3(384), 0, stream>>>(Arw, cur, T1, nxt, T2);
    _Float16* t = cur; cur = nxt; nxt = t;
  }
  k_emit<<<dim3((Bn * Cn * Hn * Wn + 255) / 256), dim3(256), 0, stream>>>(T2, (float*)d_out);
}
